// QEffMxfp4GptOssExperts_16269336117479
// MI455X (gfx1250) — matrix operand order repaired and checked by reading, not compiled, not run on hardware
//
#include <hip/hip_runtime.h>
#include <hip/hip_fp16.h>

typedef __attribute__((ext_vector_type(16))) _Float16     v16h;
typedef __attribute__((ext_vector_type(8)))  _Float16     v8h;
typedef __attribute__((ext_vector_type(8)))  float        v8f;
typedef __attribute__((ext_vector_type(4)))  int          v4i;
typedef __attribute__((ext_vector_type(8)))  int          v8i;
typedef __attribute__((ext_vector_type(4)))  unsigned int v4u;

#define ALPHA 1.702f
#define LIMIT 7.0f

// Problem constants: E=8, H=I=2880, T=1024, groups G=90, R=2I=5760
#define EXP   8
#define HDIM  2880
#define IDIM  2880
#define TTOK  1024
#define GRP   90
#define R2I   5760

// LDS A-panel: 128 rows x 32 halves, padded to 80B (40 halves) per row -> no bank conflicts
#define AP_ROWH 40
#define AP_HALF (128 * AP_ROWH)

__constant__ float FP4LUT[16] = {
    0.0f, 0.5f, 1.0f, 1.5f, 2.0f, 3.0f, 4.0f, 6.0f,
   -0.0f,-0.5f,-1.0f,-1.5f,-2.0f,-3.0f,-4.0f,-6.0f};

// ---- fp32 -> f16 activation cast --------------------------------------------
__global__ void xcast_kernel(const float* __restrict__ x, _Float16* __restrict__ xh, int n) {
    int i = blockIdx.x * 256 + threadIdx.x;
    if (i < n) xh[i] = (_Float16)x[i];
}

// ---- TDM: async 2D tile load (128 rows x 32 f16) global -> LDS --------------
// D# group0/group1 packed per CDNA5 ISA 8.3/8.4. Padding: after every 16 DWORDs
// (one 64B row) insert 4 DWORDs (16B) -> 80B LDS row stride.
__device__ __forceinline__ void tdm_load_panel(unsigned lds_addr, const void* gptr,
                                               unsigned tdim0, unsigned stride0) {
    unsigned long long ga = (unsigned long long)(uintptr_t)gptr;
    v4u g0;
    g0[0] = 1u;                                            // count=1 (valid user D#)
    g0[1] = lds_addr;                                      // LDS byte address
    g0[2] = (unsigned)(ga & 0xFFFFFFFFu);                  // global_addr[31:0]
    g0[3] = (unsigned)((ga >> 32) & 0x1FFFFFFu) | (2u << 30); // addr[56:32] | type=2
    v8i g1;
    g1[0] = (1 << 16) | (1 << 20) | (3 << 22) | (3 << 25); // data_size=2B, pad_en,
                                                           // pad_interval=16DW, pad=4DW
    g1[1] = (int)((tdim0 & 0xFFFFu) << 16);                // tensor_dim0 lo16
    g1[2] = (int)((tdim0 >> 16) & 0xFFFFu) | (128 << 16);  // tensor_dim0 hi16, tensor_dim1=128
    g1[3] = (int)(32u << 16);                              // tile_dim0 = 32 elements
    g1[4] = 128;                                           // tile_dim1 = 128 rows
    g1[5] = (int)stride0;                                  // tensor_dim0_stride lo32
    g1[6] = 0;                                             // stride hi, dim1_stride lo
    g1[7] = 0;
    v4i z4 = {0, 0, 0, 0};
    v8i z8 = {0, 0, 0, 0, 0, 0, 0, 0};
    __builtin_amdgcn_tensor_load_to_lds(g0, g1, z4, z4, z8, 0);
}

// ---- dequantize 16 consecutive K values of one MXFP4 column into a B fragment
__device__ __forceinline__ v16h dq16(const int* __restrict__ bp, float scale) {
    v4i w0 = *(const v4i*)bp;
    v4i w1 = *(const v4i*)(bp + 4);
    int by[8];
#pragma unroll
    for (int i = 0; i < 4; ++i) { by[i] = w0[i]; by[4 + i] = w1[i]; }
    v16h r;
#pragma unroll
    for (int i = 0; i < 8; ++i) {
        unsigned c = (unsigned)by[i] & 0xFFu;
        r[2 * i]     = (_Float16)(FP4LUT[c & 0xF] * scale);
        r[2 * i + 1] = (_Float16)(FP4LUT[(c >> 4) & 0xF] * scale);
    }
    return r;
}

// ---- A fragment (16x32 f16) from the padded LDS panel -----------------------
// lanes 0-15 hold K {0..15}; lanes 16-31 hold K {16..31}
__device__ __forceinline__ v16h ldsA(const _Float16* lrow, int m, int lane) {
    const int n  = lane & 15;
    const int hi = (lane & 16) ? 16 : 0;
    const _Float16* p = lrow + (m * 16 + n) * AP_ROWH + hi;
    v8h a0 = *(const v8h*)p;
    v8h a1 = *(const v8h*)(p + 8);
    v16h a;
#pragma unroll
    for (int i = 0; i < 8; ++i) { a[i] = a0[i]; a[8 + i] = a1[i]; }
    return a;
}

// ---- pass 1: gate_up GEMM + bias + GLU -> act[E,T,I] (f16) ------------------
// grid: (45, 8, 8), block 256 (8 waves). Each block: 128 tokens x 128 features.
__global__ void __launch_bounds__(256) moe_gateup_kernel(
    const _Float16* __restrict__ xh,    // [T,H] f16
    const int*      __restrict__ blk,   // [E,R2I,G,16]
    const int*      __restrict__ scl,   // [E,R2I,G]
    const float*    __restrict__ bias,  // [E,R2I]
    _Float16*       __restrict__ act)   // [E,T,I] f16
{
    __shared__ _Float16 apanel[2][AP_HALF];
    const int lane  = threadIdx.x & 31;
    const int wave  = threadIdx.x >> 5;
    const int ntile = blockIdx.x * 8 + wave;    // 0..359
    const int mbase = blockIdx.y * 8;
    const int e     = blockIdx.z;
    const int n     = lane & 15;
    const int f     = ntile * 16 + n;
    const int b0    = (lane & 16) ? 8 : 0;
    const int mofs  = (lane & 16) ? 8 : 0;

    const unsigned lds[2] = { (unsigned)(uintptr_t)(void*)&apanel[0][0],
                              (unsigned)(uintptr_t)(void*)&apanel[1][0] };
    const _Float16* abase = xh + (size_t)(blockIdx.y * 128) * HDIM;

    const int* brow = blk + (long)((e * R2I + f) * GRP) * 16;
    const int* srow = scl + (e * R2I + f) * GRP;

    v8f acc[8];
#pragma unroll
    for (int m = 0; m < 8; ++m) acc[m] = (v8f){};

    if (wave == 0) tdm_load_panel(lds[0], abase, HDIM, HDIM);

    for (int kb = 0; kb < HDIM; kb += 32) {
        const int g   = kb >> 5;
        const int cur = g & 1;
        if (wave == 0) __builtin_amdgcn_s_wait_tensorcnt(0);
        __syncthreads();
        if (wave == 0 && kb + 32 < HDIM)
            tdm_load_panel(lds[1 - cur], abase + kb + 32, HDIM, HDIM);

        __builtin_prefetch(brow + (g + 1) * 16 + b0, 0, 1);   // global_prefetch_b8
        const float scale = exp2f((float)(srow[g] - 127));
        const v16h b = dq16(brow + g * 16 + b0, scale);
        const _Float16* lrow = &apanel[cur][0];
#pragma unroll
        for (int m = 0; m < 8; ++m) {
            v16h a = ldsA(lrow, m, lane);
            acc[m] = __builtin_amdgcn_wmma_f32_16x16x32_f16(
                false, a, false, b, (short)0, acc[m], false, false);
        }
    }

    const float bf  = bias[e * R2I + f];
    const int  iout = ntile * 8 + (n >> 1);
#pragma unroll
    for (int m = 0; m < 8; ++m) {
#pragma unroll
        for (int v = 0; v < 8; ++v) {
            float d     = acc[m][v] + bf;
            float other = __shfl_xor(d, 1, 32);   // pair gate<->up columns
            float gate  = (n & 1) ? other : d;
            float up    = (n & 1) ? d : other;
            gate = fminf(gate, LIMIT);
            up   = fminf(fmaxf(up, -LIMIT), LIMIT);
            float glu = gate / (1.0f + __expf(-gate * ALPHA));
            float aa  = (up + 1.0f) * glu;
            if ((lane & 1) == 0) {
                int t = (mbase + m) * 16 + v + mofs;
                act[(long)((e << 10) + t) * IDIM + iout] = (_Float16)aa;
            }
        }
    }
}

// ---- pass 2: down GEMM + bias, routing-weighted expert sum -> out -----------
// grid: (45, 8), block 128 (4 waves). Flattened (expert,kgroup) TDM pipeline.
__global__ void __launch_bounds__(128) moe_down_kernel(
    const _Float16* __restrict__ act,   // [E,T,I] f16
    const int*      __restrict__ blk,   // [E,H,G,16]
    const int*      __restrict__ scl,   // [E,H,G]
    const float*    __restrict__ bias,  // [E,H]
    const float*    __restrict__ rw,    // [T,E]
    float*          __restrict__ out)   // [T,H]
{
    __shared__ _Float16 apanel[2][AP_HALF];
    const int lane  = threadIdx.x & 31;
    const int wave  = threadIdx.x >> 5;
    const int ntile = blockIdx.x * 4 + wave;    // 0..179
    const int mbase = blockIdx.y * 8;
    const int n     = lane & 15;
    const int h     = ntile * 16 + n;
    const int b0    = (lane & 16) ? 8 : 0;
    const int mofs  = (lane & 16) ? 8 : 0;
    const int row0  = blockIdx.y * 128;

    const unsigned lds[2] = { (unsigned)(uintptr_t)(void*)&apanel[0][0],
                              (unsigned)(uintptr_t)(void*)&apanel[1][0] };

    float oacc[8][8];
#pragma unroll
    for (int m = 0; m < 8; ++m)
#pragma unroll
        for (int v = 0; v < 8; ++v) oacc[m][v] = 0.0f;

    v8f acc[8];
#pragma unroll
    for (int m = 0; m < 8; ++m) acc[m] = (v8f){};

    if (wave == 0)
        tdm_load_panel(lds[0], act + (size_t)row0 * IDIM, IDIM, IDIM);

    for (int step = 0; step < EXP * GRP; ++step) {
        const int e   = step / GRP;
        const int g   = step % GRP;
        const int cur = step & 1;
        if (wave == 0) __builtin_amdgcn_s_wait_tensorcnt(0);
        __syncthreads();
        if (wave == 0 && step + 1 < EXP * GRP) {
            const int e2 = (step + 1) / GRP;
            const int g2 = (step + 1) % GRP;
            tdm_load_panel(lds[1 - cur],
                           act + ((size_t)(e2 << 10) + row0) * IDIM + g2 * 32,
                           IDIM, IDIM);
        }

        const int* brow = blk + (long)(((e * HDIM + h) * GRP + g)) * 16;
        const int* srow = scl + (e * HDIM + h) * GRP + g;
        __builtin_prefetch(brow + 16 + b0, 0, 1);
        const float scale = exp2f((float)(srow[0] - 127));
        const v16h b = dq16(brow + b0, scale);
        const _Float16* lrow = &apanel[cur][0];
#pragma unroll
        for (int m = 0; m < 8; ++m) {
            v16h a = ldsA(lrow, m, lane);
            acc[m] = __builtin_amdgcn_wmma_f32_16x16x32_f16(
                false, a, false, b, (short)0, acc[m], false, false);
        }

        if (g == GRP - 1) {                       // fold this expert into output acc
            const float bh = bias[e * HDIM + h];
#pragma unroll
            for (int m = 0; m < 8; ++m) {
#pragma unroll
                for (int v = 0; v < 8; ++v) {
                    int t = (mbase + m) * 16 + v + mofs;
                    oacc[m][v] += (acc[m][v] + bh) * rw[t * EXP + e];
                }
                acc[m] = (v8f){};
            }
        }
    }

#pragma unroll
    for (int m = 0; m < 8; ++m)
#pragma unroll
        for (int v = 0; v < 8; ++v) {
            int t = (mbase + m) * 16 + v + mofs;
            out[(long)t * HDIM + h] = oacc[m][v];
        }
}

extern "C" void kernel_launch(void* const* d_in, const int* in_sizes, int n_in,
                              void* d_out, int out_size, void* d_ws, size_t ws_size,
                              hipStream_t stream) {
    (void)in_sizes; (void)n_in; (void)out_size; (void)ws_size;
    const float* x      = (const float*)d_in[0];
    const float* rw     = (const float*)d_in[1];
    const int*   gub    = (const int*)d_in[2];
    const int*   gus    = (const int*)d_in[3];
    const float* gubias = (const float*)d_in[4];
    const int*   db     = (const int*)d_in[5];
    const int*   dscl   = (const int*)d_in[6];
    const float* dbias  = (const float*)d_in[7];
    float*       out    = (float*)d_out;

    _Float16* xh  = (_Float16*)d_ws;                  // T*H f16   (~5.9 MB)
    _Float16* act = xh + (size_t)TTOK * HDIM;         // E*T*I f16 (~47.2 MB)

    const int nx = TTOK * HDIM;
    xcast_kernel<<<(nx + 255) / 256, 256, 0, stream>>>(x, xh, nx);
    moe_gateup_kernel<<<dim3(45, 8, 8), 256, 0, stream>>>(xh, gub, gus, gubias, act);
    moe_down_kernel<<<dim3(45, 8), 128, 0, stream>>>(act, db, dscl, dbias, rw, out);
}
